// QuaternionTransformerBlock_43619687858907
// MI455X (gfx1250) — matrix operand order repaired and checked by reading, not compiled, not run on hardware
//
#include <hip/hip_runtime.h>

// ---------------------------------------------------------------------------
// Quaternion Transformer Block for MI455X (gfx1250), bf16 WMMA pipeline.
// All matmuls lowered to v_wmma_f32_16x16x32_bf16 (fp32 accumulate).
// Tile staging uses GLOBAL_LOAD_ASYNC_TO_LDS_B128 (ASYNCcnt) with LDS
// double buffering when the toolchain exposes the builtin.
// ---------------------------------------------------------------------------

typedef __attribute__((ext_vector_type(16))) __bf16 bf16x16;
typedef __attribute__((ext_vector_type(8)))  float  f32x8;
typedef __attribute__((ext_vector_type(4)))  int    int4v;

union FragU { uint4 u[2]; bf16x16 v; };

#if defined(__HIP_DEVICE_COMPILE__) && __has_builtin(__builtin_amdgcn_global_load_async_to_lds_b128)
#define ASYNC_LDS 1
#else
#define ASYNC_LDS 0
#endif

__device__ __forceinline__ void copy16_g2l(__bf16* lds, const __bf16* g) {
#if ASYNC_LDS
  // Per-lane async DMA: 16B global -> 16B LDS, tracked by ASYNCcnt.
  __builtin_amdgcn_global_load_async_to_lds_b128(
      (__attribute__((address_space(1))) int4v*)(g),
      (__attribute__((address_space(3))) int4v*)(lds), 0, 0);
#else
  *reinterpret_cast<uint4*>(lds) = *reinterpret_cast<const uint4*>(g);
#endif
}

__device__ __forceinline__ void wait_async0() {
#if ASYNC_LDS
#if __has_builtin(__builtin_amdgcn_s_wait_asynccnt)
  __builtin_amdgcn_s_wait_asynccnt(0);
#else
  asm volatile("s_wait_asynccnt 0x0" ::: "memory");
#endif
#endif
}

__constant__ int   c_IDX[4][4] = {{0,1,2,3},{1,0,3,2},{2,3,0,1},{3,2,1,0}};
__constant__ float c_SGN[4][4] = {{1.f,-1.f,-1.f,-1.f},
                                  {1.f, 1.f, 1.f,-1.f},
                                  {1.f,-1.f, 1.f, 1.f},
                                  {1.f, 1.f,-1.f, 1.f}};

// ------------------------- elementwise helpers ------------------------------

__device__ __forceinline__ void qmul4(const float q1[4], const float q2[4], float o[4]) {
  o[0] = q1[0]*q2[0] - q1[1]*q2[1] - q1[2]*q2[2] - q1[3]*q2[3];
  o[1] = q1[0]*q2[1] + q1[1]*q2[0] + q1[2]*q2[3] - q1[3]*q2[2];
  o[2] = q1[0]*q2[2] - q1[1]*q2[3] + q1[2]*q2[0] + q1[3]*q2[1];
  o[3] = q1[0]*q2[3] + q1[1]*q2[2] - q1[2]*q2[1] + q1[3]*q2[0];
}

// Rotation positional encoding: y = r_k * (r_ij * x), per (b,s,f) quaternion.
__global__ __launch_bounds__(256)
void rot_encode_kernel(const float* __restrict__ x, float* __restrict__ xr_f,
                       __bf16* __restrict__ xr_b, int S, int E, long long total) {
  long long idx = (long long)blockIdx.x * 256 + threadIdx.x;
  if (idx >= total) return;
  int f = (int)(idx % E);
  int s = (int)((idx / E) % S);
  float scale = __expf(-((float)f / (float)E) * 9.21034037198f);   // ln(10000)
  float theta = (float)s * (3.14159265358979323846f / (float)S) * scale;
  float sn, cs; __sincosf(theta, &sn, &cs);
  const float4 xv = *reinterpret_cast<const float4*>(x + idx * 4);
  float q[4]  = {xv.x, xv.y, xv.z, xv.w};
  float rij[4] = {cs, -sn,  sn, cs};
  float rk[4]  = {cs,  sn, -sn, cs};
  float t[4], y[4];
  qmul4(rij, q, t);
  qmul4(rk,  t, y);
  *reinterpret_cast<float4*>(xr_f + idx * 4) = make_float4(y[0], y[1], y[2], y[3]);
  xr_b[idx*4+0] = (__bf16)y[0];
  xr_b[idx*4+1] = (__bf16)y[1];
  xr_b[idx*4+2] = (__bf16)y[2];
  xr_b[idx*4+3] = (__bf16)y[3];
}

// Expand quaternion weight stack w(4,out,in) into real bf16 matrix
// W[(g*4+c) * (4*in) + (f*4+d)] = SGN[c][d] * w[IDX[c][d], g, f]
__global__ __launch_bounds__(256)
void expand_w_kernel(const float* __restrict__ w, __bf16* __restrict__ W,
                     int outF, int inF, long long total) {
  long long idx = (long long)blockIdx.x * 256 + threadIdx.x;
  if (idx >= total) return;
  int cols = 4 * inF;
  long long row = idx / cols;
  int col = (int)(idx % cols);
  int c = (int)(row & 3); int g = (int)(row >> 2);
  int d = col & 3;        int f = col >> 2;
  int comp = c_IDX[c][d];
  float val = c_SGN[c][d] * w[(size_t)comp * outF * inF + (size_t)g * inF + f];
  W[idx] = (__bf16)val;
}

// Row softmax (ncols = 1024, 256 threads, 4 elems/thread); in-place fp32 +
// bf16 copy for the attn @ V GEMM.
__global__ __launch_bounds__(256)
void softmax_rows_kernel(float* __restrict__ sc, __bf16* __restrict__ attn_b, int ncols) {
  long long row = blockIdx.x;
  float* r = sc + row * (long long)ncols;
  __shared__ float red[256];
  float v[4];
  float mx = -1e30f;
  #pragma unroll
  for (int i = 0; i < 4; ++i) { v[i] = r[threadIdx.x + i*256]; mx = fmaxf(mx, v[i]); }
  red[threadIdx.x] = mx; __syncthreads();
  for (int s2 = 128; s2 > 0; s2 >>= 1) {
    if (threadIdx.x < (unsigned)s2) red[threadIdx.x] = fmaxf(red[threadIdx.x], red[threadIdx.x + s2]);
    __syncthreads();
  }
  mx = red[0]; __syncthreads();
  float sum = 0.f;
  #pragma unroll
  for (int i = 0; i < 4; ++i) { v[i] = __expf(v[i] - mx); sum += v[i]; }
  red[threadIdx.x] = sum; __syncthreads();
  for (int s2 = 128; s2 > 0; s2 >>= 1) {
    if (threadIdx.x < (unsigned)s2) red[threadIdx.x] += red[threadIdx.x + s2];
    __syncthreads();
  }
  float inv = 1.f / red[0];
  #pragma unroll
  for (int i = 0; i < 4; ++i) {
    float a = v[i] * inv;
    r[threadIdx.x + i*256] = a;
    attn_b[row * (long long)ncols + threadIdx.x + i*256] = (__bf16)a;
  }
}

// modReLU on (b,s,h,4) quaternions: scale = relu(|q| + bias[h]) / (|q|+1e-6)
__global__ __launch_bounds__(256)
void modrelu_kernel(const float* __restrict__ h, const float* __restrict__ bias,
                    __bf16* __restrict__ out_b, int H, long long total) {
  long long idx = (long long)blockIdx.x * 256 + threadIdx.x;
  if (idx >= total) return;
  float4 q = *reinterpret_cast<const float4*>(h + idx * 4);
  float nrm = sqrtf(q.x*q.x + q.y*q.y + q.z*q.z + q.w*q.w);
  float sc = fmaxf(nrm + bias[idx % H], 0.f) / (nrm + 1e-6f);
  out_b[idx*4+0] = (__bf16)(q.x * sc);
  out_b[idx*4+1] = (__bf16)(q.y * sc);
  out_b[idx*4+2] = (__bf16)(q.z * sc);
  out_b[idx*4+3] = (__bf16)(q.w * sc);
}

// y = xin + add; out = gamma * y / sqrt(|y|^2 + eps) + beta  (per quaternion)
__global__ __launch_bounds__(256)
void resid_ln_kernel(const float* __restrict__ xin, const float* __restrict__ addv,
                     const float* __restrict__ gamma, const float* __restrict__ beta,
                     float* __restrict__ outf, __bf16* __restrict__ outb,
                     int feat, long long total) {
  long long idx = (long long)blockIdx.x * 256 + threadIdx.x;
  if (idx >= total) return;
  float4 a = *reinterpret_cast<const float4*>(xin  + idx * 4);
  float4 b = *reinterpret_cast<const float4*>(addv + idx * 4);
  float y0 = a.x + b.x, y1 = a.y + b.y, y2 = a.z + b.z, y3 = a.w + b.w;
  float nrm = sqrtf(y0*y0 + y1*y1 + y2*y2 + y3*y3 + 1e-6f);
  float inv = 1.f / nrm;
  int f = (int)(idx % feat);
  float4 g  = *reinterpret_cast<const float4*>(gamma + (size_t)f * 4);
  float4 bt = *reinterpret_cast<const float4*>(beta  + (size_t)f * 4);
  float o0 = g.x * y0 * inv + bt.x;
  float o1 = g.y * y1 * inv + bt.y;
  float o2 = g.z * y2 * inv + bt.z;
  float o3 = g.w * y3 * inv + bt.w;
  *reinterpret_cast<float4*>(outf + idx * 4) = make_float4(o0, o1, o2, o3);
  if (outb) {
    outb[idx*4+0] = (__bf16)o0; outb[idx*4+1] = (__bf16)o1;
    outb[idx*4+2] = (__bf16)o2; outb[idx*4+3] = (__bf16)o3;
  }
}

// ------------------------- bf16 WMMA GEMM -----------------------------------
// C = A(M x K) * B(N x K)^T, A/B bf16 row-major (K-major), fp32 accumulate.
// Block tile 128x128x32, 256 threads = 8 wave32; wave tile 32x64 = 2x4 WMMA.
// Double-buffered LDS; staging via async global->LDS DMA when available.

enum { EPI_BF16 = 0, EPI_BF16T = 1, EPI_F32_SCALE = 2, EPI_F32_BIAS = 3 };

template <int EPI>
__global__ __launch_bounds__(256)
void gemm_bf16_wmma(const __bf16* __restrict__ A, const __bf16* __restrict__ Bm,
                    const float* __restrict__ bias, float scale,
                    float* __restrict__ Cf, __bf16* __restrict__ Cb,
                    int K, int ldc,
                    long long sA, long long sB, long long sC) {
  A  += (size_t)blockIdx.z * sA;
  Bm += (size_t)blockIdx.z * sB;
  if constexpr (EPI == EPI_F32_SCALE || EPI == EPI_F32_BIAS) Cf += (size_t)blockIdx.z * sC;
  else                                                       Cb += (size_t)blockIdx.z * sC;

  // +8 bf16 pad: 80B row stride (16B aligned, bank-conflict friendly).
  __shared__ __bf16 As[2][128][40];
  __shared__ __bf16 Bs[2][128][40];

  const int tid  = threadIdx.x;
  const int lane = tid & 31;
  const int wave = tid >> 5;
  const int hl   = lane >> 4;       // half-wave: 0 or 1
  const int r16  = lane & 15;
  const int wm   = wave >> 1;       // 0..3  (row quadrant)
  const int wn   = wave & 1;        // 0..1  (col half)

  const int rowBase = blockIdx.y * 128;
  const int colBase = blockIdx.x * 128;

  // Staging map: 128x32 bf16 per operand = 512 16B chunks, 2 per thread.
  const int r0  = tid >> 2;               // chunk row (first chunk)
  const int kc0 = (tid & 3) * 8;          // chunk col offset

  f32x8 acc[2][4];
  #pragma unroll
  for (int i = 0; i < 2; ++i)
    #pragma unroll
    for (int j = 0; j < 4; ++j)
      #pragma unroll
      for (int e = 0; e < 8; ++e) acc[i][j][e] = 0.f;

  auto stage = [&](int buf, int k0) {
    copy16_g2l(&As[buf][r0][kc0],      A  + (size_t)(rowBase + r0)      * K + k0 + kc0);
    copy16_g2l(&Bs[buf][r0][kc0],      Bm + (size_t)(colBase + r0)      * K + k0 + kc0);
    copy16_g2l(&As[buf][r0 + 64][kc0], A  + (size_t)(rowBase + r0 + 64) * K + k0 + kc0);
    copy16_g2l(&Bs[buf][r0 + 64][kc0], Bm + (size_t)(colBase + r0 + 64) * K + k0 + kc0);
  };

  const int nIter = K >> 5;   // K / 32
  stage(0, 0);
  wait_async0();
  __syncthreads();

  for (int it = 0; it < nIter; ++it) {
    const int cur = it & 1;
    if (it + 1 < nIter) stage(1 - cur, (it + 1) << 5);

    // A fragment (16x32): lanes 0-15 hold k {0-7,16-23}; lanes 16-31 {8-15,24-31}.
    bf16x16 af[2];
    #pragma unroll
    for (int i = 0; i < 2; ++i) {
      int row = wm * 32 + i * 16 + r16;
      FragU u;
      u.u[0] = *reinterpret_cast<const uint4*>(&As[cur][row][hl * 8]);
      u.u[1] = *reinterpret_cast<const uint4*>(&As[cur][row][hl * 8 + 16]);
      af[i] = u.v;
    }
    // B fragment (32x16): lanes 0-15 hold k {0-7,16-23}; lanes 16-31 {8-15,24-31}.
    bf16x16 bfr[4];
    #pragma unroll
    for (int j = 0; j < 4; ++j) {
      int row = wn * 64 + j * 16 + r16;
      FragU u;
      u.u[0] = *reinterpret_cast<const uint4*>(&Bs[cur][row][hl * 8]);
      u.u[1] = *reinterpret_cast<const uint4*>(&Bs[cur][row][hl * 8 + 16]);
      bfr[j] = u.v;
    }
    #pragma unroll
    for (int i = 0; i < 2; ++i)
      #pragma unroll
      for (int j = 0; j < 4; ++j)
        acc[i][j] = __builtin_amdgcn_wmma_f32_16x16x32_bf16(
            false, af[i], false, bfr[j], (short)0, acc[i][j], false, false);

    // Own async copies must land before anyone consumes the next buffer.
    wait_async0();
    __syncthreads();
  }

  // Epilogue. C layout: n = r16 (+wave/block col offsets), m = e + 8*hl.
  #pragma unroll
  for (int i = 0; i < 2; ++i) {
    int mBase = rowBase + wm * 32 + i * 16 + hl * 8;
    #pragma unroll
    for (int j = 0; j < 4; ++j) {
      int n = colBase + wn * 64 + j * 16 + r16;
      float bval = 0.f;
      if constexpr (EPI == EPI_BF16 || EPI == EPI_BF16T || EPI == EPI_F32_BIAS)
        bval = bias[n];
      #pragma unroll
      for (int e = 0; e < 8; ++e) {
        int m = mBase + e;
        float val = acc[i][j][e];
        if constexpr (EPI == EPI_F32_SCALE)      Cf[(size_t)m * ldc + n] = val * scale;
        else if constexpr (EPI == EPI_F32_BIAS)  Cf[(size_t)m * ldc + n] = val + bval;
        else if constexpr (EPI == EPI_BF16)      Cb[(size_t)m * ldc + n] = (__bf16)(val + bval);
        else /* EPI_BF16T */                     Cb[(size_t)n * ldc + m] = (__bf16)(val + bval);
      }
    }
  }
}

// ------------------------------- launch -------------------------------------

extern "C" void kernel_launch(void* const* d_in, const int* in_sizes, int n_in,
                              void* d_out, int out_size, void* d_ws, size_t ws_size,
                              hipStream_t stream) {
  (void)in_sizes; (void)n_in; (void)out_size; (void)ws_size;

  constexpr int Bb = 16, Ss = 1024, Ee = 256, Hh = 1024;
  constexpr int FE = 4 * Ee;        // 1024
  constexpr int FH = 4 * Hh;        // 4096
  constexpr long long Mrows = (long long)Bb * Ss;          // 16384
  constexpr size_t nX    = (size_t)Mrows * FE;             // 16.78M elems
  constexpr size_t nH    = (size_t)Mrows * FH;             // 67.1M elems
  constexpr size_t nAttn = (size_t)Bb * Ss * Ss;           // 16.78M elems

  const float* x    = (const float*)d_in[0];
  const float* q_w  = (const float*)d_in[1];
  const float* q_b  = (const float*)d_in[2];
  const float* k_w  = (const float*)d_in[3];
  const float* k_b  = (const float*)d_in[4];
  const float* v_w  = (const float*)d_in[5];
  const float* v_b  = (const float*)d_in[6];
  const float* f1_w = (const float*)d_in[7];
  const float* f1_b = (const float*)d_in[8];
  const float* modb = (const float*)d_in[9];
  const float* f2_w = (const float*)d_in[10];
  const float* f2_b = (const float*)d_in[11];
  const float* g1   = (const float*)d_in[12];
  const float* bt1  = (const float*)d_in[13];
  const float* g2   = (const float*)d_in[14];
  const float* bt2  = (const float*)d_in[15];

  float* out_x    = (float*)d_out;
  float* out_attn = out_x + nX;     // raw scores -> softmax in place

  // Workspace bump allocator (~560 MB; dead regions aliased below).
  char* p = (char*)d_ws;
  auto alloc = [&](size_t bytes) -> void* {
    void* r = (void*)p;
    p += (bytes + 255) & ~(size_t)255;
    return r;
  };
  __bf16* Wq = (__bf16*)alloc((size_t)FE * FE * 2);
  __bf16* Wk = (__bf16*)alloc((size_t)FE * FE * 2);
  __bf16* Wv = (__bf16*)alloc((size_t)FE * FE * 2);
  __bf16* W1 = (__bf16*)alloc((size_t)FH * FE * 2);
  __bf16* W2 = (__bf16*)alloc((size_t)FE * FH * 2);
  char* span = p;                                  // reusable region start
  float*  xr_f   = (float*)alloc(nX * 4);
  __bf16* xr_b   = (__bf16*)alloc(nX * 2);
  __bf16* Qb     = (__bf16*)alloc(nX * 2);
  __bf16* Kb     = (__bf16*)alloc(nX * 2);
  __bf16* Vt     = (__bf16*)alloc(nX * 2);         // V^T per batch: [b][n][s]
  __bf16* attnB  = (__bf16*)alloc(nAttn * 2);
  float*  attnO  = (float*)alloc(nX * 4);
  float*  x1_f   = (float*)alloc(nX * 4);
  __bf16* x1_b   = (__bf16*)alloc(nX * 2);
  __bf16* h1_b   = (__bf16*)alloc(nH * 2);
  // Aliases into dead regions (lifetimes disjoint):
  float* h1_f  = (float*)span;   // 268.4 MB over dead xr..attnO span (302 MB)
  float* ff2_f = (float*)span;   // 67.1 MB, after h1_f is consumed

  const dim3 blk(256);
  auto ew = [](long long n) { return dim3((unsigned)((n + 255) / 256)); };

  // 1) Expand quaternion weights to real bf16 matrices.
  expand_w_kernel<<<ew(16LL*Ee*Ee), blk, 0, stream>>>(q_w,  Wq, Ee, Ee, 16LL*Ee*Ee);
  expand_w_kernel<<<ew(16LL*Ee*Ee), blk, 0, stream>>>(k_w,  Wk, Ee, Ee, 16LL*Ee*Ee);
  expand_w_kernel<<<ew(16LL*Ee*Ee), blk, 0, stream>>>(v_w,  Wv, Ee, Ee, 16LL*Ee*Ee);
  expand_w_kernel<<<ew(16LL*Hh*Ee), blk, 0, stream>>>(f1_w, W1, Hh, Ee, 16LL*Hh*Ee);
  expand_w_kernel<<<ew(16LL*Ee*Hh), blk, 0, stream>>>(f2_w, W2, Ee, Hh, 16LL*Ee*Hh);

  // 2) Rotation positional encoding (fp32 for residual + bf16 for GEMM).
  rot_encode_kernel<<<ew(Mrows*Ee), blk, 0, stream>>>(x, xr_f, xr_b, Ss, Ee, Mrows*Ee);

  // 3) Q/K projections (bf16 out); V projection stored transposed per batch.
  gemm_bf16_wmma<EPI_BF16><<<dim3(FE/128, (unsigned)(Mrows/128), 1), blk, 0, stream>>>(
      xr_b, Wq, q_b, 1.f, nullptr, Qb, FE, FE, 0, 0, 0);
  gemm_bf16_wmma<EPI_BF16><<<dim3(FE/128, (unsigned)(Mrows/128), 1), blk, 0, stream>>>(
      xr_b, Wk, k_b, 1.f, nullptr, Kb, FE, FE, 0, 0, 0);
  gemm_bf16_wmma<EPI_BF16T><<<dim3(FE/128, Ss/128, Bb), blk, 0, stream>>>(
      xr_b, Wv, v_b, 1.f, nullptr, Vt, FE, Ss,
      (long long)Ss * FE, 0, (long long)FE * Ss);

  // 4) scores = Q K^T / 32, batched; written straight into d_out attn region.
  gemm_bf16_wmma<EPI_F32_SCALE><<<dim3(Ss/128, Ss/128, Bb), blk, 0, stream>>>(
      Qb, Kb, nullptr, 1.f / 32.f, out_attn, nullptr, FE, Ss,
      (long long)Ss * FE, (long long)Ss * FE, (long long)Ss * Ss);

  // 5) softmax in place (+ bf16 copy for the next GEMM).
  softmax_rows_kernel<<<dim3((unsigned)(Bb * Ss)), blk, 0, stream>>>(out_attn, attnB, Ss);

  // 6) attn_out = attn @ V (B operand = V^T, K-major).
  gemm_bf16_wmma<EPI_F32_SCALE><<<dim3(FE/128, Ss/128, Bb), blk, 0, stream>>>(
      attnB, Vt, nullptr, 1.f, attnO, nullptr, Ss, FE,
      (long long)Ss * Ss, (long long)FE * Ss, (long long)Ss * FE);

  // 7) x1 = qlayernorm(x_rot + attn_out)
  resid_ln_kernel<<<ew(Mrows*Ee), blk, 0, stream>>>(xr_f, attnO, g1, bt1,
                                                    x1_f, x1_b, Ee, Mrows*Ee);

  // 8) FFN: h1 = x1 @ W1^T + b1 ; modReLU ; ff2 = h1 @ W2^T + b2
  gemm_bf16_wmma<EPI_F32_BIAS><<<dim3(FH/128, (unsigned)(Mrows/128), 1), blk, 0, stream>>>(
      x1_b, W1, f1_b, 1.f, h1_f, nullptr, FE, FH, 0, 0, 0);
  modrelu_kernel<<<ew(Mrows*Hh), blk, 0, stream>>>(h1_f, modb, h1_b, Hh, Mrows*Hh);
  gemm_bf16_wmma<EPI_F32_BIAS><<<dim3(FE/128, (unsigned)(Mrows/128), 1), blk, 0, stream>>>(
      h1_b, W2, f2_b, 1.f, ff2_f, nullptr, FH, FE, 0, 0, 0);

  // 9) out = qlayernorm(x1 + ff2)
  resid_ln_kernel<<<ew(Mrows*Ee), blk, 0, stream>>>(x1_f, ff2_f, g2, bt2,
                                                    out_x, nullptr, Ee, Mrows*Ee);
}
